// Flow_9740985827713
// MI455X (gfx1250) — matrix operand order repaired and checked by reading, not compiled, not run on hardware
//
#include <hip/hip_runtime.h>
#include <hip/hip_bf16.h>
#include <stdint.h>
#include <math.h>

// ---------------------------------------------------------------------------
// RQS normalizing-flow log_prob on gfx1250.
// GEMMs via v_wmma_f32_16x16x32_f16 (D = W^T * X^T, so C-frag -> LDS -> B-frag
// needs no cross-lane transpose). Weights pre-swizzled to A-fragment layout.
// ---------------------------------------------------------------------------

typedef _Float16 v16h __attribute__((ext_vector_type(16)));
typedef float    v8f  __attribute__((ext_vector_type(8)));

#define NBINS   10
#define MIN_WW  1e-3f
#define MIN_HH  1e-3f
#define MIN_DD  1e-3f
#define BATCH   65536
#define WAVES   2          // waves per workgroup (wave32); 16 samples per wave

union Frag {               // one lane's 32-byte slice of a 16x32 f16 fragment
  v16h h;
  int4 i[2];
  _Float16 e[16];
};
union Half8 { _Float16 e[8]; int4 i; };
union F4    { float4 v; float f[4]; };

// ---------------------------------------------------------------------------
// Prologue: f32 weights -> f16, padded + swizzled into A-fragment byte order.
// A-frag (16x32 f16, M=n rows, K=k cols), per ISA 7.12.2:
//   lane la: row n = la&15, khi = la>>4
//   vgpr v, half hf: k = ((v&4)<<2) + khi*8 + (v&3)*2 + hf
// Memory image: frag*1024B + la*32B + v*4B + hf*2B  (lane-major, 2x b128/lane)
// ---------------------------------------------------------------------------
__global__ void swizzle_weights(const float* __restrict__ src,
                                _Float16* __restrict__ dst,
                                int K_src, int N_src, int KT,
                                int src_blk_stride, int dst_blk_stride_h,
                                int per_blk) {
  int tid = blockIdx.x * blockDim.x + threadIdx.x;
  if (tid >= per_blk * 8) return;
  int blk = tid / per_blk, r = tid - blk * per_blk;
  int frag = r >> 9, e = r & 511;                 // 512 halves per fragment
  int la = e >> 4, q = e & 15, v = q >> 1, hf = q & 1;
  int nt = frag / KT, kt = frag - nt * KT;
  int nr = la & 15, khi = la >> 4;
  int kc = ((v & 4) << 2) + khi * 8 + ((v & 3) << 1) + hf;
  int n = nt * 16 + nr, k = kt * 32 + kc;
  float val = (k < K_src && n < N_src) ? src[blk * src_blk_stride + k * N_src + n]
                                       : 0.0f;
  dst[blk * dst_blk_stride_h + r] = (_Float16)val;
}

// ---------------------------------------------------------------------------
// One coupling block: MLP (wmma) + RQS spline + logdet accumulation.
// Each wave owns 16 samples, a private 8KB f16 activation strip hT[16][256]
// and a private 16KB f32 param strip p[16][256]. No barriers needed.
// ---------------------------------------------------------------------------
__launch_bounds__(WAVES * 32)
__global__ void flow_block(float* __restrict__ xw,
                           const _Float16* __restrict__ wA,  // this block's swizzled weights
                           const float* __restrict__ b0,
                           const float* __restrict__ b1,
                           const float* __restrict__ b2,
                           const float* __restrict__ b3,
                           float* __restrict__ jac,
                           int accumulate,
                           unsigned cond_p, unsigned trafo_p) {
  __shared__ _Float16 lds_h[WAVES * 4096];  // hT: [16 samples][256 feats] f16 per wave
  __shared__ float    lds_p[WAVES * 4096];  // p : [16 samples][256 feats] f32 per wave

  const int lane = threadIdx.x & 31;
  const int wave = threadIdx.x >> 5;
  const int m    = lane & 15;               // sample column within tile
  const int hi   = lane >> 4;
  const int sbase = (blockIdx.x * WAVES + wave) * 16;
  const int s     = sbase + m;

  _Float16* hT = lds_h + wave * 4096;
  float*    pT = lds_p + wave * 4096;

  auto loadA = [&](const _Float16* base, int fragIdx) {
    Frag f;
    const int4* p = reinterpret_cast<const int4*>(base + fragIdx * 512 + lane * 16);
    f.i[0] = p[0]; f.i[1] = p[1];
    return f;
  };
  auto loadB = [&](int kt) {                // B-frag: col m, K = kt*32 + hi*8 .. +7, kt*32 + 16 + hi*8 .. +7
    Frag f;
    const int4* p = reinterpret_cast<const int4*>(hT + m * 256 + kt * 32 + hi * 8);
    f.i[0] = p[0]; f.i[1] = p[2];
    return f;
  };
  auto storeH = [&](const v8f& acc, const float* bias, int nt) {
    int nb = nt * 16 + hi * 8;              // C-frag: lane holds n = nb..nb+7, col m
    F4 bv0, bv1;
    bv0.v = *reinterpret_cast<const float4*>(bias + nb);
    bv1.v = *reinterpret_cast<const float4*>(bias + nb + 4);
    Half8 h8;
#pragma unroll
    for (int r = 0; r < 8; ++r) {
      float b = (r < 4) ? bv0.f[r] : bv1.f[r - 4];
      h8.e[r] = (_Float16)fmaxf(acc[r] + b, 0.0f);
    }
    *reinterpret_cast<int4*>(hT + m * 256 + nb) = h8.i;   // one ds_store_b128
  };

  // ---- layer 0: build B-frag = x_cond^T (K: 8 real, padded to 32) ----------
  Frag bx;
#pragma unroll
  for (int j = 0; j < 16; ++j) bx.e[j] = (_Float16)0.0f;
  if (hi == 0) {
#pragma unroll
    for (int j = 0; j < 4; ++j) {           // vgpr j holds k=2j,2j+1 (k<8 real)
      int d0 = (cond_p >> (8 * j))     & 15;
      int d1 = (cond_p >> (8 * j + 4)) & 15;
      bx.e[2 * j]     = (_Float16)xw[s * 16 + d0];
      bx.e[2 * j + 1] = (_Float16)xw[s * 16 + d1];
    }
  }
#pragma unroll 4
  for (int nt = 0; nt < 16; ++nt) {
    Frag a = loadA(wA, nt);                 // w0^T frags (16 frags, KT=1)
    v8f acc = {};
    acc = __builtin_amdgcn_wmma_f32_16x16x32_f16(false, a.h, false, bx.h,
                                                 (short)0, acc, false, false);
    storeH(acc, b0, nt);
  }

  // ---- layers 1 & 2: 256x256 -----------------------------------------------
  auto layer = [&](const _Float16* wbase, const float* bias) {
    Frag B[8];
#pragma unroll
    for (int kt = 0; kt < 8; ++kt) B[kt] = loadB(kt);  // frees hT for overwrite
#pragma unroll 2
    for (int nt = 0; nt < 16; ++nt) {
      v8f acc = {};
#pragma unroll
      for (int kt = 0; kt < 8; ++kt) {
        Frag a = loadA(wbase, nt * 8 + kt);
        acc = __builtin_amdgcn_wmma_f32_16x16x32_f16(false, a.h, false, B[kt].h,
                                                     (short)0, acc, false, false);
      }
      storeH(acc, bias, nt);
    }
  };
  layer(wA + 8192,  b1);
  layer(wA + 73728, b2);

  // ---- layer 3: 256 -> 248 (padded 256), f32 params to LDS -----------------
  {
    Frag B[8];
#pragma unroll
    for (int kt = 0; kt < 8; ++kt) B[kt] = loadB(kt);
#pragma unroll 2
    for (int nt = 0; nt < 16; ++nt) {
      v8f acc = {};
#pragma unroll
      for (int kt = 0; kt < 8; ++kt) {
        Frag a = loadA(wA + 139264, nt * 8 + kt);
        acc = __builtin_amdgcn_wmma_f32_16x16x32_f16(false, a.h, false, B[kt].h,
                                                     (short)0, acc, false, false);
      }
      int nb = nt * 16 + hi * 8;
      float4 v0, v1;
      v0.x = acc[0]; v0.y = acc[1]; v0.z = acc[2]; v0.w = acc[3];
      v1.x = acc[4]; v1.y = acc[5]; v1.z = acc[6]; v1.w = acc[7];
      *reinterpret_cast<float4*>(pT + m * 256 + nb)     = v0;
      *reinterpret_cast<float4*>(pT + m * 256 + nb + 4) = v1;
    }
  }

  // ---- rational-quadratic spline: 16 samples x 8 dims = 128 tasks ----------
#pragma unroll 1
  for (int it = 0; it < 4; ++it) {
    int id = it * 32 + lane;
    int sm = id >> 3;                       // sample in tile
    int t  = id & 7;                        // transformed-dim slot
    int ss = sbase + sm;
    int td = (trafo_p >> (4 * t)) & 15;
    const float* pp  = pT + sm * 256 + t * 31;
    const float* bp  = b3 + t * 31;
    float P[31];
#pragma unroll
    for (int j = 0; j < 31; ++j) P[j] = pp[j] + bp[j];

    float xv = xw[ss * 16 + td];
    bool inside = (xv >= 0.0f) && (xv <= 1.0f);
    float xc = fminf(fmaxf(xv, 0.0f), 1.0f);

    // widths softmax -> cumulative edges c[0..10]
    float mw = P[0];
#pragma unroll
    for (int j = 1; j < NBINS; ++j) mw = fmaxf(mw, P[j]);
    float ew[NBINS]; float se = 0.0f;
#pragma unroll
    for (int j = 0; j < NBINS; ++j) { ew[j] = expf(P[j] - mw); se += ew[j]; }
    float c[NBINS + 1]; c[0] = 0.0f;
    float scw = (1.0f - MIN_WW * NBINS) / se;
#pragma unroll
    for (int j = 0; j < NBINS; ++j) c[j + 1] = c[j] + (MIN_WW + ew[j] * scw);

    // heights softmax -> cumulative edges cE[0..10]
    float mh = P[NBINS];
#pragma unroll
    for (int j = 1; j < NBINS; ++j) mh = fmaxf(mh, P[NBINS + j]);
    float eh[NBINS]; float seh = 0.0f;
#pragma unroll
    for (int j = 0; j < NBINS; ++j) { eh[j] = expf(P[NBINS + j] - mh); seh += eh[j]; }
    float cE[NBINS + 1]; cE[0] = 0.0f;
    float sch = (1.0f - MIN_HH * NBINS) / seh;
#pragma unroll
    for (int j = 0; j < NBINS; ++j) cE[j + 1] = cE[j] + (MIN_HH + eh[j] * sch);

    // derivatives
    float dd[NBINS + 1];
#pragma unroll
    for (int j = 0; j <= NBINS; ++j) {
      float u = P[2 * NBINS + j];
      float spv = (u > 20.0f) ? u : log1pf(expf(u));
      dd[j] = MIN_DD + spv;
    }

    // bin select (running cndmask; c[0]=0 <= xc always)
    float cwk = c[0], cwk1 = c[1], chk = cE[0], chk1 = cE[1], dk = dd[0], dk1 = dd[1];
#pragma unroll
    for (int j = 1; j < NBINS; ++j) {
      bool sel = (c[j] <= xc);
      cwk  = sel ? c[j]      : cwk;   cwk1 = sel ? c[j + 1]  : cwk1;
      chk  = sel ? cE[j]     : chk;   chk1 = sel ? cE[j + 1] : chk1;
      dk   = sel ? dd[j]     : dk;    dk1  = sel ? dd[j + 1] : dk1;
    }
    float wk = cwk1 - cwk, hk = chk1 - chk;
    float th  = (xc - cwk) / wk;
    float t1m = th * (1.0f - th);
    float delta = hk / wk;
    float den = delta + (dk + dk1 - 2.0f * delta) * t1m;
    float y   = chk + hk * (delta * th * th + dk * t1m) / den;
    float ld  = 2.0f * logf(delta)
              + logf(dk1 * th * th + 2.0f * delta * t1m + dk * (1.0f - th) * (1.0f - th))
              - 2.0f * logf(den);
    y  = inside ? y  : xv;
    ld = inside ? ld : 0.0f;

    xw[ss * 16 + td] = y;

    // sum logdet over the 8 dims of this sample (8 consecutive lanes)
    ld += __shfl_xor(ld, 1, 32);
    ld += __shfl_xor(ld, 2, 32);
    ld += __shfl_xor(ld, 4, 32);
    if (t == 0) {
      float prev = accumulate ? jac[ss] : 0.0f;
      jac[ss] = prev + ld;
    }
  }
}

// ---------------------------------------------------------------------------
extern "C" void kernel_launch(void* const* d_in, const int* in_sizes, int n_in,
                              void* d_out, int out_size, void* d_ws, size_t ws_size,
                              hipStream_t stream) {
  const float* x  = (const float*)d_in[0];
  const float* w0 = (const float*)d_in[1];
  const float* b0 = (const float*)d_in[2];
  const float* w1 = (const float*)d_in[3];
  const float* b1 = (const float*)d_in[4];
  const float* w2 = (const float*)d_in[5];
  const float* b2 = (const float*)d_in[6];
  const float* w3 = (const float*)d_in[7];
  const float* b3 = (const float*)d_in[8];
  float* jac = (float*)d_out;

  // workspace: [ x_work: 65536*16 f32 | swizzled f16 weights: 8 * 204800 halves ]
  float*    xwork = (float*)d_ws;
  _Float16* wA    = (_Float16*)((char*)d_ws + (size_t)BATCH * 16 * 4);
  const int DSTRIDE = 204800;   // halves per coupling block (16KB+3*128KB bytes)

  hipMemcpyAsync(xwork, x, (size_t)BATCH * 16 * sizeof(float),
                 hipMemcpyDeviceToDevice, stream);

  // swizzle each layer's weights (all 8 blocks per launch)
  {
    int per = 16 * 1 * 512, total = per * 8;   // L0: K 8->32, KT=1
    swizzle_weights<<<(total + 255) / 256, 256, 0, stream>>>(
        w0, wA + 0, 8, 256, 1, 8 * 256, DSTRIDE, per);
  }
  {
    int per = 16 * 8 * 512, total = per * 8;   // L1
    swizzle_weights<<<(total + 255) / 256, 256, 0, stream>>>(
        w1, wA + 8192, 256, 256, 8, 256 * 256, DSTRIDE, per);
  }
  {
    int per = 16 * 8 * 512, total = per * 8;   // L2
    swizzle_weights<<<(total + 255) / 256, 256, 0, stream>>>(
        w2, wA + 73728, 256, 256, 8, 256 * 256, DSTRIDE, per);
  }
  {
    int per = 16 * 8 * 512, total = per * 8;   // L3: N 248->256
    swizzle_weights<<<(total + 255) / 256, 256, 0, stream>>>(
        w3, wA + 139264, 256, 248, 8, 256 * 248, DSTRIDE, per);
  }

  // 8 sequential coupling blocks ('log' permutation masks computed on host)
  for (int blk = 0; blk < 8; ++blk) {
    int bbit = blk >> 1, inv = blk & 1;
    unsigned cond_p = 0, trafo_p = 0;
    int ci = 0, ti = 0;
    for (int d = 0; d < 16; ++d) {
      int mk = ((d >> bbit) & 1) ^ inv;     // True = conditioning dim
      if (mk) cond_p  |= (unsigned)d << (4 * ci++);
      else    trafo_p |= (unsigned)d << (4 * ti++);
    }
    flow_block<<<BATCH / (WAVES * 16), WAVES * 32, 0, stream>>>(
        xwork, wA + (size_t)blk * DSTRIDE,
        b0 + blk * 256, b1 + blk * 256, b2 + blk * 256, b3 + blk * 248,
        jac, (blk != 0) ? 1 : 0, cond_p, trafo_p);
  }
}
